// DLSM_42666205118483
// MI455X (gfx1250) — compile-verified, matrix operand order checked by reading, not run on hardware
//
#include <hip/hip_runtime.h>
#include <hip/hip_bf16.h>
#include <stdint.h>

// ---------------------------------------------------------------------------
// GraphSAGE-style 2-layer encoder for MI455X (gfx1250, wave32, WMMA).
// Aggregations in fp32 (L2-resident gathers), GEMMs via v_wmma_f32_16x16x32_bf16.
// ---------------------------------------------------------------------------

typedef __bf16 bf16_t;
typedef __attribute__((ext_vector_type(16))) __bf16 v16bf;
typedef __attribute__((ext_vector_type(8)))  __bf16 v8bf;
typedef __attribute__((ext_vector_type(8)))  float  v8f;

#define BATCH      256
#define FDIM       128     // feature / embedding width
#define DDIM       64      // final projection width
#define MAXDEG     64
#define N_HOP0     25      // NUM_NEIGHBORS[1]
#define N_HOP1     10      // NUM_NEIGHBORS[0]
#define S1_W       (2 * N_HOP0)            // 50
#define P_HOP1     (BATCH * S1_W)          // 12800
#define KDIM       (3 * FDIM)              // 384

// ---------------------------------------------------------------------------
// 1) Deterministic shared column permutations (Fisher-Yates, xorshift32).
//    perm id p = set*4 + hop*2 + dir ; hop 0 -> n=25, hop 1 -> n=10.
//    Stored at cols + p*32.
// ---------------------------------------------------------------------------
__global__ void init_cols_kernel(int* __restrict__ cols) {
    int p = threadIdx.x;
    if (p >= 8) return;
    unsigned st = 0x9E3779B9u * (unsigned)(p + 1) ^ 0x0000C0DEu;
    int perm[MAXDEG];
    for (int i = 0; i < MAXDEG; ++i) perm[i] = i;
    for (int i = MAXDEG - 1; i > 0; --i) {
        st ^= st << 13; st ^= st >> 17; st ^= st << 5;
        int j = (int)(st % (unsigned)(i + 1));
        int t = perm[i]; perm[i] = perm[j]; perm[j] = t;
    }
    int n = ((p >> 1) & 1) ? N_HOP1 : N_HOP0;
    for (int i = 0; i < n; ++i) cols[p * 32 + i] = perm[i];
}

// ---------------------------------------------------------------------------
// 2) Hop-0 sample list: s1[r*50 + j] = neighbors_{out,in}[nodes[r]][cols[j]]
// ---------------------------------------------------------------------------
__global__ void build_s1_kernel(const int* __restrict__ nodes,
                                const int* __restrict__ nbr_out,
                                const int* __restrict__ nbr_in,
                                const int* __restrict__ cols_out,
                                const int* __restrict__ cols_in,
                                int* __restrict__ s1) {
    int r = blockIdx.x;
    int j = threadIdx.x;
    if (j >= S1_W) return;
    int node = nodes[r];
    int v = (j < N_HOP0)
        ? nbr_out[(size_t)node * MAXDEG + cols_out[j]]
        : nbr_in [(size_t)node * MAXDEG + cols_in [j - N_HOP0]];
    s1[(size_t)r * S1_W + j] = v;
}

// ---------------------------------------------------------------------------
// 3a) Hop-0 aggregation: A[r] = bf16( feat[self] || mean_out(25) || mean_in(25) )
//     nei indices come from the precomputed s1 rows. 1 block / row, 128 thr.
// ---------------------------------------------------------------------------
__global__ void agg_hop0_kernel(const int* __restrict__ self_idx,
                                const int* __restrict__ s1,
                                const float* __restrict__ feat,
                                bf16_t* __restrict__ A) {
    int r = blockIdx.x;
    int f = threadIdx.x;                      // 0..127
    float sv = feat[(size_t)self_idx[r] * FDIM + f];
    const int* row = s1 + (size_t)r * S1_W;
    float so = 0.f, si = 0.f;
    for (int j = 0; j < N_HOP0; ++j) so += feat[(size_t)row[j] * FDIM + f];
    for (int j = 0; j < N_HOP0; ++j) si += feat[(size_t)row[N_HOP0 + j] * FDIM + f];
    bf16_t* out = A + (size_t)r * KDIM;
    out[f]            = (bf16_t)sv;
    out[FDIM + f]     = (bf16_t)(so * (1.0f / N_HOP0));
    out[2 * FDIM + f] = (bf16_t)(si * (1.0f / N_HOP0));
}

// ---------------------------------------------------------------------------
// 3b) Hop-1 aggregation (the big one, P=12800): neighbor indices computed on
//     the fly from the neighbor tables + shared column permutation (fused s2).
// ---------------------------------------------------------------------------
__global__ void agg_hop1_kernel(const int* __restrict__ s1,          // self idx
                                const int* __restrict__ nbr_out,
                                const int* __restrict__ nbr_in,
                                const int* __restrict__ cols_out,
                                const int* __restrict__ cols_in,
                                const float* __restrict__ feat,
                                bf16_t* __restrict__ A) {
    int r = blockIdx.x;
    int f = threadIdx.x;
    int s = s1[r];
    float sv = feat[(size_t)s * FDIM + f];
    const int* no = nbr_out + (size_t)s * MAXDEG;
    const int* ni = nbr_in  + (size_t)s * MAXDEG;
    float so = 0.f, si = 0.f;
    for (int j = 0; j < N_HOP1; ++j) so += feat[(size_t)no[cols_out[j]] * FDIM + f];
    for (int j = 0; j < N_HOP1; ++j) si += feat[(size_t)ni[cols_in [j]] * FDIM + f];
    bf16_t* out = A + (size_t)r * KDIM;
    out[f]            = (bf16_t)sv;
    out[FDIM + f]     = (bf16_t)(so * (1.0f / N_HOP1));
    out[2 * FDIM + f] = (bf16_t)(si * (1.0f / N_HOP1));
}

// ---------------------------------------------------------------------------
// 3c) Layer-1 aggregation over layer-0 activations (shared by all 3 heads):
//     A[r] = bf16( h0[r] || mean_{j<25} h1[r*50+j] || mean_{j<25} h1[r*50+25+j] )
// ---------------------------------------------------------------------------
__global__ void agg_h_kernel(const float* __restrict__ h_self,   // [256][128]
                             const float* __restrict__ h_nei,    // [12800][128]
                             bf16_t* __restrict__ A) {            // [256][384]
    int r = blockIdx.x;
    int f = threadIdx.x;
    float so = 0.f, si = 0.f;
    for (int j = 0; j < N_HOP0; ++j) so += h_nei[((size_t)r * S1_W + j) * FDIM + f];
    for (int j = 0; j < N_HOP0; ++j) si += h_nei[((size_t)r * S1_W + N_HOP0 + j) * FDIM + f];
    bf16_t* out = A + (size_t)r * KDIM;
    out[f]            = (bf16_t)h_self[(size_t)r * FDIM + f];
    out[FDIM + f]     = (bf16_t)(so * (1.0f / N_HOP0));
    out[2 * FDIM + f] = (bf16_t)(si * (1.0f / N_HOP0));
}

// ---------------------------------------------------------------------------
// 4) Weight transpose + fp32->bf16:  src[K][N] -> dst[N][K]
// ---------------------------------------------------------------------------
__global__ void tconv_kernel(const float* __restrict__ src,
                             bf16_t* __restrict__ dst, int K, int N) {
    int idx = blockIdx.x * blockDim.x + threadIdx.x;
    if (idx >= K * N) return;
    int k = idx / N, n = idx % N;
    dst[(size_t)n * K + k] = (bf16_t)src[(size_t)k * N + n];
}

// ---------------------------------------------------------------------------
// 5) WMMA GEMM:  C = act(A[M][K] * B[K][N]),  B given transposed (Bt[N][K]).
//    One wave -> one 16x16 output tile; K-loop in steps of 32 with
//    v_wmma_f32_16x16x32_bf16.  A/B fragments per the CDNA5 16-bit layouts:
//    lane<16: row lane&15, K = 0..7 & 16..23 ; lane>=16: K = 8..15 & 24..31.
//    All addresses 16B-aligned (K multiple of 32, buffers 256B-aligned).
// ---------------------------------------------------------------------------
__global__ void gemm_wmma_kernel(const bf16_t* __restrict__ A,   // [M][K]
                                 const bf16_t* __restrict__ Bt,  // [N][K]
                                 float*  __restrict__ Cf,        // [M][N] or null
                                 bf16_t* __restrict__ Cb,        // [M][N] or null
                                 int M, int N, int K, int act_sigmoid) {
    int wavesPerBlk = blockDim.x >> 5;
    int wave = blockIdx.x * wavesPerBlk + (threadIdx.x >> 5);
    int tilesN = N >> 4;
    int totalTiles = (M >> 4) * tilesN;
    if (wave >= totalTiles) return;                 // uniform per wave
    int tm = wave / tilesN;
    int tn = wave % tilesN;
    int lane  = threadIdx.x & 31;
    int mrow  = (tm << 4) + (lane & 15);
    int nrow  = (tn << 4) + (lane & 15);
    int kbase = (lane >> 4) << 3;                   // 0 or 8

    const bf16_t* Arow = A  + (size_t)mrow * K + kbase;
    const bf16_t* Brow = Bt + (size_t)nrow * K + kbase;

    v8f c = {};
    for (int kk = 0; kk < K; kk += 32) {
        v8bf a0 = *(const v8bf*)(Arow + kk);
        v8bf a1 = *(const v8bf*)(Arow + kk + 16);
        v8bf b0 = *(const v8bf*)(Brow + kk);
        v8bf b1 = *(const v8bf*)(Brow + kk + 16);
        v16bf a = __builtin_shufflevector(a0, a1, 0,1,2,3,4,5,6,7,8,9,10,11,12,13,14,15);
        v16bf b = __builtin_shufflevector(b0, b1, 0,1,2,3,4,5,6,7,8,9,10,11,12,13,14,15);
        c = __builtin_amdgcn_wmma_f32_16x16x32_bf16(
                /*neg_a=*/false, a, /*neg_b=*/false, b,
                /*c_mod=*/(short)0, c, /*reuse_a=*/false, /*reuse_b=*/false);
    }

    // C/D layout: VGPR r -> M = r (+8 for lanes 16..31), N = lane&15.
    int mbase = (tm << 4) + ((lane >> 4) << 3);
    int ncol  = (tn << 4) + (lane & 15);
#pragma unroll
    for (int r = 0; r < 8; ++r) {
        float x = c[r];
        if (act_sigmoid) x = 1.0f / (1.0f + __expf(-x));
        size_t off = (size_t)(mbase + r) * N + ncol;
        if (Cf) Cf[off] = x;
        if (Cb) Cb[off] = (bf16_t)x;
    }
}

// ---------------------------------------------------------------------------
// Host-side orchestration
// ---------------------------------------------------------------------------
extern "C" void kernel_launch(void* const* d_in, const int* in_sizes, int n_in,
                              void* d_out, int out_size, void* d_ws, size_t ws_size,
                              hipStream_t stream) {
    const int*   nodes[2]  = { (const int*)d_in[0], (const int*)d_in[1] };
    const int*   nbr_out   = (const int*)d_in[2];
    const int*   nbr_in    = (const int*)d_in[3];
    const float* feat      = (const float*)d_in[4];
    const float* W_in      = (const float*)d_in[5];
    const float* W_head[3] = { (const float*)d_in[6], (const float*)d_in[7], (const float*)d_in[8] };
    const float* Wd_head[3]= { (const float*)d_in[11], (const float*)d_in[12], (const float*)d_in[13] };
    float* out = (float*)d_out;

    // -------- workspace carve-out (256B aligned) --------
    char* wsb = (char*)d_ws;
    size_t off = 0;
    auto carve = [&](size_t bytes) -> void* {
        off = (off + 255) & ~(size_t)255;
        void* p = wsb + off;
        off += bytes;
        return p;
    };
    int*    cols     = (int*)carve(8 * 32 * sizeof(int));
    int*    s1[2];
    bf16_t* A_hop1[2]; bf16_t* A_hop0[2]; bf16_t* A_l1[2];
    float*  H_hop1[2]; float*  H_hop0[2];
    bf16_t* Hb_head[2][3];
    for (int s = 0; s < 2; ++s) {
        s1[s]     = (int*)   carve((size_t)P_HOP1 * sizeof(int));
        A_hop1[s] = (bf16_t*)carve((size_t)P_HOP1 * KDIM * sizeof(bf16_t));
        A_hop0[s] = (bf16_t*)carve((size_t)BATCH  * KDIM * sizeof(bf16_t));
        A_l1[s]   = (bf16_t*)carve((size_t)BATCH  * KDIM * sizeof(bf16_t));
        H_hop1[s] = (float*) carve((size_t)P_HOP1 * FDIM * sizeof(float));
        H_hop0[s] = (float*) carve((size_t)BATCH  * FDIM * sizeof(float));
        for (int h = 0; h < 3; ++h)
            Hb_head[s][h] = (bf16_t*)carve((size_t)BATCH * FDIM * sizeof(bf16_t));
    }
    bf16_t* WinT = (bf16_t*)carve((size_t)FDIM * KDIM * sizeof(bf16_t));
    bf16_t* WheadT[3]; bf16_t* WdT[3];
    for (int h = 0; h < 3; ++h) {
        WheadT[h] = (bf16_t*)carve((size_t)FDIM * KDIM * sizeof(bf16_t));
        WdT[h]    = (bf16_t*)carve((size_t)DDIM * FDIM * sizeof(bf16_t));
    }

    // -------- 1) permutations + weight conversion --------
    init_cols_kernel<<<1, 8, 0, stream>>>(cols);
    {
        int elems = KDIM * FDIM;                 // 384*128
        int blks  = (elems + 255) / 256;
        tconv_kernel<<<blks, 256, 0, stream>>>(W_in, WinT, KDIM, FDIM);
        for (int h = 0; h < 3; ++h)
            tconv_kernel<<<blks, 256, 0, stream>>>(W_head[h], WheadT[h], KDIM, FDIM);
        int elemsd = FDIM * DDIM;                // 128*64
        int blksd  = (elemsd + 255) / 256;
        for (int h = 0; h < 3; ++h)
            tconv_kernel<<<blksd, 256, 0, stream>>>(Wd_head[h], WdT[h], FDIM, DDIM);
    }

    const int GEMM_WPB = 4;                      // waves per block
    const int GEMM_BLK = 32 * GEMM_WPB;

    for (int s = 0; s < 2; ++s) {
        const int* c25o = cols + (s * 4 + 0) * 32;
        const int* c25i = cols + (s * 4 + 1) * 32;
        const int* c10o = cols + (s * 4 + 2) * 32;
        const int* c10i = cols + (s * 4 + 3) * 32;

        // -------- 2) sampling + layer-0 aggregation --------
        build_s1_kernel<<<BATCH, 64, 0, stream>>>(nodes[s], nbr_out, nbr_in, c25o, c25i, s1[s]);
        agg_hop0_kernel<<<BATCH, FDIM, 0, stream>>>(nodes[s], s1[s], feat, A_hop0[s]);
        agg_hop1_kernel<<<P_HOP1, FDIM, 0, stream>>>(s1[s], nbr_out, nbr_in, c10o, c10i, feat, A_hop1[s]);

        // -------- 3) layer-0 GEMMs (sigmoid) --------
        {
            int tiles = (BATCH / 16) * (FDIM / 16);            // 128
            gemm_wmma_kernel<<<(tiles + GEMM_WPB - 1) / GEMM_WPB, GEMM_BLK, 0, stream>>>(
                A_hop0[s], WinT, H_hop0[s], (bf16_t*)nullptr, BATCH, FDIM, KDIM, 1);
        }
        {
            int tiles = (P_HOP1 / 16) * (FDIM / 16);           // 6400
            gemm_wmma_kernel<<<(tiles + GEMM_WPB - 1) / GEMM_WPB, GEMM_BLK, 0, stream>>>(
                A_hop1[s], WinT, H_hop1[s], (bf16_t*)nullptr, P_HOP1, FDIM, KDIM, 1);
        }

        // -------- 4) layer-1 aggregation (shared by heads) --------
        agg_h_kernel<<<BATCH, FDIM, 0, stream>>>(H_hop0[s], H_hop1[s], A_l1[s]);

        // -------- 5) heads + final projections --------
        for (int h = 0; h < 3; ++h) {
            int tiles1 = (BATCH / 16) * (FDIM / 16);           // 128
            gemm_wmma_kernel<<<(tiles1 + GEMM_WPB - 1) / GEMM_WPB, GEMM_BLK, 0, stream>>>(
                A_l1[s], WheadT[h], (float*)nullptr, Hb_head[s][h], BATCH, FDIM, KDIM, 1);

            int tiles2 = (BATCH / 16) * (DDIM / 16);           // 64
            float* zout = out + (size_t)(s * 3 + h) * BATCH * DDIM;
            gemm_wmma_kernel<<<(tiles2 + GEMM_WPB - 1) / GEMM_WPB, GEMM_BLK, 0, stream>>>(
                Hb_head[s][h], WdT[h], zout, (bf16_t*)nullptr, BATCH, DDIM, FDIM, 0);
        }
    }
}
